// ConfigurableMamba_32349693673989
// MI455X (gfx1250) — compile-verified, matrix operand order checked by reading, not run on hardware
//
#include <hip/hip_runtime.h>
#include <hip/hip_bf16.h>
#include <math.h>

// ---------------- config (matches reference) ----------------
#define NL 4
#define D_MODEL 256
#define N_CH 64
#define N_CLS 5
#define D_INNER 512
#define D_STATE 64
#define D_CONV 4
#define HP 64
#define NH 8
#define CONV_DIM 640        // D_INNER + 2*D_STATE
#define D_PROJ 1160         // 2*D_INNER + 2*D_STATE + NH
#define BATCH 16
#define SEQ 2048
#define NTOK (BATCH*SEQ)    // 32768
#define EPS 1e-5f

// ---------------- WMMA plumbing (CDNA5 wave32) ----------------
typedef __attribute__((ext_vector_type(16))) __bf16 v16bf;
typedef __attribute__((ext_vector_type(8)))  float  v8f;

union F8   { v8f   v; float  f[8]; };
union BF16 { v16bf v; __bf16 e[16]; uint4 q[2]; };
union BF4  { __bf16 h[4]; unsigned long long u; };

__device__ __forceinline__ v8f wmma_bf16(v16bf a, v16bf b, v8f c) {
  // D = A(16x32 bf16) x B(32x16 bf16) + C(16x16 f32)
  return __builtin_amdgcn_wmma_f32_16x16x32_bf16(
      /*neg_a=*/false, a, /*neg_b=*/false, b,
      /*c_mod=*/(short)0, c, /*reuse_a=*/false, /*reuse_b=*/false);
}

// Fragment from K-contiguous (row-major [idx x K]) bf16 LDS buffer.
// element K=k taken from p[(idx0+lane%16)*ld + k0 + k].
// Base offset is a multiple of 16B by construction -> two ds_load_b128.
__device__ __forceinline__ v16bf frag_row(const __bf16* p, int ld, int idx0, int k0) {
  int lane = threadIdx.x & 31;
  const __bf16* base = p + (idx0 + (lane & 15)) * ld + k0 + ((lane >> 4) * 8);
  BF16 f;
  f.q[0] = *(const uint4*)(base);        // K = kh+0..7   (pairs packed per VGPR)
  f.q[1] = *(const uint4*)(base + 16);   // K = kh+16..23
  return f.v;
}

__device__ __forceinline__ float block_reduce128(float v, float* sred, int tid) {
  __syncthreads();               // protect prior use of sred
  sred[tid] = v; __syncthreads();
#pragma unroll
  for (int s = 64; s > 0; s >>= 1) {
    if (tid < s) sred[tid] += sred[tid + s];
    __syncthreads();
  }
  return sred[0];
}

// ---------------- generic GEMM: C = A(MxK) * B(KxN) [+ bias] ----------------
// block = 128 threads (4 wave32), tile 64x64, K step 64, bf16 WMMA / f32 acc.
// Double-buffered LDS; all fragment loads for a tile are issued before the
// 8-WMMA chain so LDS waits amortize over the whole chain.
__device__ __forceinline__ void gemm_gload(
    const float* __restrict__ A, const float* __restrict__ B,
    int N, int K, int m0, int n0, int kk, int tid,
    float4 ra[8], float4 rb[8]) {
#pragma unroll
  for (int j = 0; j < 8; ++j) {
    int q = j * 128 + tid;
    int r = q >> 4, c4 = (q & 15) * 4;
    ra[j] = *(const float4*)&A[(size_t)(m0 + r) * K + kk + c4];
  }
  if (n0 + 64 <= N) {
#pragma unroll
    for (int j = 0; j < 8; ++j) {
      int q = j * 128 + tid;
      int r = q >> 4, c4 = (q & 15) * 4;
      rb[j] = *(const float4*)&B[(size_t)(kk + r) * N + n0 + c4];
    }
  } else {
#pragma unroll
    for (int j = 0; j < 8; ++j) {
      int q = j * 128 + tid;
      int r = q >> 4, c4 = (q & 15) * 4;
      const float* row = &B[(size_t)(kk + r) * N];
      float4 v;
      v.x = (n0 + c4 + 0 < N) ? row[n0 + c4 + 0] : 0.f;
      v.y = (n0 + c4 + 1 < N) ? row[n0 + c4 + 1] : 0.f;
      v.z = (n0 + c4 + 2 < N) ? row[n0 + c4 + 2] : 0.f;
      v.w = (n0 + c4 + 3 < N) ? row[n0 + c4 + 3] : 0.f;
      rb[j] = v;
    }
  }
}

__device__ __forceinline__ void gemm_sstore(
    __bf16* sA, __bf16* sBT, int tid,
    const float4 ra[8], const float4 rb[8]) {
#pragma unroll
  for (int j = 0; j < 8; ++j) {
    int q = j * 128 + tid;
    int r = q >> 4, c4 = (q & 15) * 4;
    BF4 t; t.h[0] = (__bf16)ra[j].x; t.h[1] = (__bf16)ra[j].y;
           t.h[2] = (__bf16)ra[j].z; t.h[3] = (__bf16)ra[j].w;
    *(unsigned long long*)&sA[r * 64 + c4] = t.u;        // ds_store_b64
    sBT[(c4 + 0) * 64 + r] = (__bf16)rb[j].x;            // transpose on store
    sBT[(c4 + 1) * 64 + r] = (__bf16)rb[j].y;
    sBT[(c4 + 2) * 64 + r] = (__bf16)rb[j].z;
    sBT[(c4 + 3) * 64 + r] = (__bf16)rb[j].w;
  }
}

__global__ void __launch_bounds__(128) k_gemm(
    const float* __restrict__ A, const float* __restrict__ B,
    float* __restrict__ C, const float* __restrict__ bias,
    int M, int N, int K) {
  __shared__ alignas(16) __bf16 sA [2][64 * 64];   // [m][k]
  __shared__ alignas(16) __bf16 sBT[2][64 * 64];   // [n][k]  (B transposed)
  int tid  = threadIdx.x;
  int lane = tid & 31, wave = tid >> 5;
  int m0 = blockIdx.y * 64, n0 = blockIdx.x * 64;

  const v8f zero8 = {0.f,0.f,0.f,0.f,0.f,0.f,0.f,0.f};
  v8f acc[4] = {zero8, zero8, zero8, zero8};

  float4 ra[8], rb[8];
  gemm_gload(A, B, N, K, m0, n0, 0, tid, ra, rb);
  gemm_sstore(sA[0], sBT[0], tid, ra, rb);
  __syncthreads();

  int cur = 0;
  for (int kk = 0; kk < K; kk += 64) {
    bool more = (kk + 64 < K);
    if (more)                                  // issue next tile's loads early
      gemm_gload(A, B, N, K, m0, n0, kk + 64, tid, ra, rb);

    // hoist ALL fragment loads, then run the 8-WMMA chain back-to-back
    v16bf af0 = frag_row(sA[cur], 64, wave * 16, 0);
    v16bf af1 = frag_row(sA[cur], 64, wave * 16, 32);
    v16bf bf0[4], bf1[4];
#pragma unroll
    for (int nt = 0; nt < 4; ++nt) {
      bf0[nt] = frag_row(sBT[cur], 64, nt * 16, 0);
      bf1[nt] = frag_row(sBT[cur], 64, nt * 16, 32);
    }
#pragma unroll
    for (int nt = 0; nt < 4; ++nt) acc[nt] = wmma_bf16(af0, bf0[nt], acc[nt]);
#pragma unroll
    for (int nt = 0; nt < 4; ++nt) acc[nt] = wmma_bf16(af1, bf1[nt], acc[nt]);

    if (more) {
      gemm_sstore(sA[cur ^ 1], sBT[cur ^ 1], tid, ra, rb);
      __syncthreads();
      cur ^= 1;
    }
  }

  int half = lane >> 4, cl = lane & 15;
#pragma unroll
  for (int nt = 0; nt < 4; ++nt) {
    F8 u; u.v = acc[nt];
    int n = n0 + nt * 16 + cl;
    if (n < N) {
      float bv = bias ? bias[n] : 0.f;
#pragma unroll
      for (int r = 0; r < 8; ++r) {
        int m = m0 + wave * 16 + r + 8 * half;
        C[(size_t)m * N + n] = u.f[r] + bv;
      }
    }
  }
}

// ---------------- depthwise causal conv1d (w=4) + bias + SiLU ----------------
__global__ void __launch_bounds__(256) k_conv_silu(
    const float* __restrict__ zxb, const float* __restrict__ cw,
    const float* __restrict__ cb, float* __restrict__ xBC) {
  int gid = blockIdx.x * 256 + threadIdx.x;
  if (gid >= NTOK * CONV_DIM) return;
  int c = gid % CONV_DIM;
  int l = (gid / CONV_DIM) % SEQ;
  int b = gid / (CONV_DIM * SEQ);
  float acc = cb[c];
#pragma unroll
  for (int j = 0; j < D_CONV; ++j) {
    int lt = l - (D_CONV - 1) + j;
    if (lt >= 0)
      acc += zxb[(size_t)(b * SEQ + lt) * D_PROJ + D_INNER + c] * cw[c * D_CONV + j];
  }
  xBC[gid] = acc / (1.f + __expf(-acc));   // SiLU
}

// ---------------- dt = softplus(dt_raw + dt_bias) ----------------
__global__ void __launch_bounds__(256) k_dt(
    const float* __restrict__ zxb, const float* __restrict__ dt_bias,
    float* __restrict__ dtv) {
  int gid = blockIdx.x * 256 + threadIdx.x;
  if (gid >= NTOK * NH) return;
  int hh = gid & (NH - 1);
  int row = gid >> 3;
  float x = zxb[(size_t)row * D_PROJ + 2 * D_INNER + 2 * D_STATE + hh] + dt_bias[hh];
  dtv[gid] = (x > 20.f) ? x : log1pf(__expf(x));
}

// ---------------- chunked SSD scan (Mamba2), one block per (b,h) ----------------
// State S[n][p] fp32 in LDS; per 64-chunk: 4 matmuls of 64x64x64 via bf16 WMMA.
// All WMMA operands staged K-contiguous: sXT=[p][m], sTmp=[p][n] (S^T snapshot,
// later reused as Bsc^T [n][m]), sB=[l][n], sC=[l][n], sG=[l][m].
// Per K-slice, the A-frag and all 4 B-frags are loaded before the 4-WMMA chain.
__global__ void __launch_bounds__(128) k_scan(
    const float* __restrict__ xBC, const float* __restrict__ dtv,
    const float* __restrict__ A_log, const float* __restrict__ Dp,
    float* __restrict__ yout) {
  __shared__ alignas(16) float  S   [64 * 64];   // state [n][p], fp32
  __shared__ alignas(16) __bf16 sTmp[64 * 64];   // S^T snapshot -> Bsc^T
  __shared__ alignas(16) __bf16 sXT [64 * 64];   // X^T  [p][l]
  __shared__ alignas(16) __bf16 sB  [64 * 64];   // B    [l][n]
  __shared__ alignas(16) __bf16 sC  [64 * 64];   // C    [l][n]
  __shared__ alignas(16) __bf16 sG  [64 * 64];   // masked (C B^T) [l][m]
  __shared__ float cum[64];                      // inclusive cumsum of dt*A
  __shared__ float dtc[64];
  __shared__ float sScale[64];                   // dt[m]*exp(cumQ-cum[m])

  int bh = blockIdx.x;
  int b = bh >> 3, h = bh & 7;
  int tid = threadIdx.x, lane = tid & 31, wave = tid >> 5;
  int half = lane >> 4, cl = lane & 15;

  const float Ah = -__expf(A_log[h]);
  const float Dh = Dp[h];
  const v8f zero8 = {0.f,0.f,0.f,0.f,0.f,0.f,0.f,0.f};

  for (int i = tid; i < 4096; i += 128) S[i] = 0.f;
  __syncthreads();

  for (int q = 0; q < SEQ / 64; ++q) {
    int l0 = q * 64;
    // ---- stage chunk: float4 global loads, batched; transpose on LDS store ----
#pragma unroll
    for (int j = 0; j < 8; ++j) {
      int qd = j * 128 + tid;                     // 1024 quads
      int l = qd >> 4, p4 = (qd & 15) * 4;        // l = token (also n for S)
      const float* rowp = &xBC[(size_t)(b * SEQ + l0 + l) * CONV_DIM];
      float4 xv = *(const float4*)&rowp[h * HP + p4];
      float4 bv = *(const float4*)&rowp[D_INNER + p4];
      float4 cv = *(const float4*)&rowp[D_INNER + D_STATE + p4];
      float4 sv = *(const float4*)&S[l * 64 + p4];
      sXT[(p4 + 0) * 64 + l] = (__bf16)xv.x;      // X^T scatter
      sXT[(p4 + 1) * 64 + l] = (__bf16)xv.y;
      sXT[(p4 + 2) * 64 + l] = (__bf16)xv.z;
      sXT[(p4 + 3) * 64 + l] = (__bf16)xv.w;
      BF4 tb; tb.h[0]=(__bf16)bv.x; tb.h[1]=(__bf16)bv.y;
              tb.h[2]=(__bf16)bv.z; tb.h[3]=(__bf16)bv.w;
      *(unsigned long long*)&sB[l * 64 + p4] = tb.u;
      BF4 tc; tc.h[0]=(__bf16)cv.x; tc.h[1]=(__bf16)cv.y;
              tc.h[2]=(__bf16)cv.z; tc.h[3]=(__bf16)cv.w;
      *(unsigned long long*)&sC[l * 64 + p4] = tc.u;
      sTmp[(p4 + 0) * 64 + l] = (__bf16)sv.x;     // S^T scatter
      sTmp[(p4 + 1) * 64 + l] = (__bf16)sv.y;
      sTmp[(p4 + 2) * 64 + l] = (__bf16)sv.z;
      sTmp[(p4 + 3) * 64 + l] = (__bf16)sv.w;
    }
    if (tid < 64) dtc[tid] = dtv[(size_t)(b * SEQ + l0 + tid) * NH + h];
    __syncthreads();
    if (tid == 0) {
      float cacc = 0.f;
      for (int l = 0; l < 64; ++l) { cacc += dtc[l] * Ah; cum[l] = cacc; }
    }
    __syncthreads();
    float cumQ = cum[63];
    if (tid < 64) sScale[tid] = dtc[tid] * __expf(cumQ - cum[tid]);

    // ---- matmul1: G = C @ B^T (rows l band per wave, K = n) ----
    v8f g[4] = {zero8, zero8, zero8, zero8};
#pragma unroll
    for (int ks = 0; ks < 64; ks += 32) {
      v16bf a = frag_row(sC, 64, wave * 16, ks);
      v16bf bb[4];
#pragma unroll
      for (int nt = 0; nt < 4; ++nt) bb[nt] = frag_row(sB, 64, nt * 16, ks);
#pragma unroll
      for (int nt = 0; nt < 4; ++nt) g[nt] = wmma_bf16(a, bb[nt], g[nt]);
    }
    // causal decay mask: G'[l,m] = G * exp(cum[l]-cum[m]) * dt[m], m<=l
#pragma unroll
    for (int nt = 0; nt < 4; ++nt) {
      F8 u; u.v = g[nt];
#pragma unroll
      for (int r = 0; r < 8; ++r) {
        int l = wave * 16 + r + 8 * half;
        int m = nt * 16 + cl;
        float val = (m <= l) ? u.f[r] * __expf(cum[l] - cum[m]) * dtc[m] : 0.f;
        sG[l * 64 + m] = (__bf16)val;
      }
    }
    __syncthreads();

    // ---- matmul2: Yintra = G' @ X ; matmul3: Z = C @ S_prev ----
    v8f y4[4] = {zero8, zero8, zero8, zero8};
    v8f z4[4] = {zero8, zero8, zero8, zero8};
#pragma unroll
    for (int ks = 0; ks < 64; ks += 32) {
      v16bf ag = frag_row(sG, 64, wave * 16, ks);
      v16bf ac = frag_row(sC, 64, wave * 16, ks);
      v16bf bx[4], bs[4];
#pragma unroll
      for (int nt = 0; nt < 4; ++nt) {
        bx[nt] = frag_row(sXT,  64, nt * 16, ks);
        bs[nt] = frag_row(sTmp, 64, nt * 16, ks);
      }
#pragma unroll
      for (int nt = 0; nt < 4; ++nt) y4[nt] = wmma_bf16(ag, bx[nt], y4[nt]);
#pragma unroll
      for (int nt = 0; nt < 4; ++nt) z4[nt] = wmma_bf16(ac, bs[nt], z4[nt]);
    }
    __syncthreads();   // S^T snapshot no longer needed

    // ---- build Bsc^T[n][m] = B[m][n]*dt[m]*exp(cumQ-cum[m]) into sTmp ----
#pragma unroll
    for (int j = 0; j < 8; ++j) {
      int qd = j * 128 + tid;
      int m = qd >> 4, n4 = (qd & 15) * 4;
      BF4 t; t.u = *(const unsigned long long*)&sB[m * 64 + n4];  // ds_load_b64
      float s = sScale[m];
      sTmp[(n4 + 0) * 64 + m] = (__bf16)((float)t.h[0] * s);
      sTmp[(n4 + 1) * 64 + m] = (__bf16)((float)t.h[1] * s);
      sTmp[(n4 + 2) * 64 + m] = (__bf16)((float)t.h[2] * s);
      sTmp[(n4 + 3) * 64 + m] = (__bf16)((float)t.h[3] * s);
    }
    __syncthreads();

    // ---- matmul4: dS[n][p] = Bsc^T @ X (rows n band per wave, K = m) ----
    v8f ds4[4] = {zero8, zero8, zero8, zero8};
#pragma unroll
    for (int ks = 0; ks < 64; ks += 32) {
      v16bf a = frag_row(sTmp, 64, wave * 16, ks);
      v16bf bb[4];
#pragma unroll
      for (int nt = 0; nt < 4; ++nt) bb[nt] = frag_row(sXT, 64, nt * 16, ks);
#pragma unroll
      for (int nt = 0; nt < 4; ++nt) ds4[nt] = wmma_bf16(a, bb[nt], ds4[nt]);
    }
    // state update + Y epilogue (D skip fused)
    float eQ = __expf(cumQ);
#pragma unroll
    for (int nt = 0; nt < 4; ++nt) {
      F8 uy, uz, us; uy.v = y4[nt]; uz.v = z4[nt]; us.v = ds4[nt];
#pragma unroll
      for (int r = 0; r < 8; ++r) {
        int rr = wave * 16 + r + 8 * half;   // = l for Y, = n for S
        int pp = nt * 16 + cl;               // = p
        S[rr * 64 + pp] = S[rr * 64 + pp] * eQ + us.f[r];
        float xv = (float)sXT[pp * 64 + rr];
        float yv = uy.f[r] + __expf(cum[rr]) * uz.f[r] + Dh * xv;
        yout[(size_t)(b * SEQ + l0 + rr) * D_INNER + h * HP + pp] = yv;
      }
    }
    __syncthreads();   // before next chunk overwrites LDS / reads S
  }
}

// ---------------- gated RMSNorm: y = rmsnorm(y * silu(z)) * w (in place) ----------------
__global__ void __launch_bounds__(128) k_gate_rms(
    const float* __restrict__ zxb, const float* __restrict__ nw,
    float* __restrict__ y) {
  __shared__ float sred[128];
  int row = blockIdx.x, tid = threadIdx.x;
  float v[4], acc = 0.f;
#pragma unroll
  for (int j = 0; j < 4; ++j) {
    int d = tid + 128 * j;
    float yy = y[(size_t)row * D_INNER + d];
    float z  = zxb[(size_t)row * D_PROJ + d];
    float t  = yy * (z / (1.f + __expf(-z)));
    v[j] = t; acc += t * t;
  }
  float ss = block_reduce128(acc, sred, tid);
  float scale = rsqrtf(ss / (float)D_INNER + EPS);
#pragma unroll
  for (int j = 0; j < 4; ++j) {
    int d = tid + 128 * j;
    y[(size_t)row * D_INNER + d] = v[j] * scale * nw[d];
  }
}

// ---------------- residual + LayerNorm: h = LN(m + h)*w + b (in place) ----------------
__global__ void __launch_bounds__(128) k_res_ln(
    const float* __restrict__ m, const float* __restrict__ lnw,
    const float* __restrict__ lnb, float* __restrict__ h) {
  __shared__ float sred[128];
  int row = blockIdx.x, tid = threadIdx.x;
  float v[2], s = 0.f;
#pragma unroll
  for (int j = 0; j < 2; ++j) {
    int d = tid + 128 * j;
    v[j] = m[(size_t)row * D_MODEL + d] + h[(size_t)row * D_MODEL + d];
    s += v[j];
  }
  float mean = block_reduce128(s, sred, tid) / (float)D_MODEL;
  float s2 = 0.f;
#pragma unroll
  for (int j = 0; j < 2; ++j) { float d0 = v[j] - mean; s2 += d0 * d0; }
  float var = block_reduce128(s2, sred, tid) / (float)D_MODEL;
  float rs = rsqrtf(var + EPS);
#pragma unroll
  for (int j = 0; j < 2; ++j) {
    int d = tid + 128 * j;
    h[(size_t)row * D_MODEL + d] = (v[j] - mean) * rs * lnw[d] + lnb[d];
  }
}

// ---------------- output projection (N=5, tiny) ----------------
__global__ void __launch_bounds__(256) k_linear_out(
    const float* __restrict__ h, const float* __restrict__ w,
    const float* __restrict__ b, float* __restrict__ out) {
  int gid = blockIdx.x * 256 + threadIdx.x;
  if (gid >= NTOK * N_CLS) return;
  int row = gid / N_CLS, c = gid % N_CLS;
  float acc = b[c];
  for (int d = 0; d < D_MODEL; ++d)
    acc += h[(size_t)row * D_MODEL + d] * w[d * N_CLS + c];
  out[gid] = acc;
}

// ---------------- driver ----------------
extern "C" void kernel_launch(void* const* d_in, const int* in_sizes, int n_in,
                              void* d_out, int out_size, void* d_ws, size_t ws_size,
                              hipStream_t stream) {
  const float* x        = (const float*)d_in[0];
  const float* lin_in_w = (const float*)d_in[1];
  const float* lin_in_b = (const float*)d_in[2];
  const float* W_in     = (const float*)d_in[3];
  const float* conv_w   = (const float*)d_in[4];
  const float* conv_b   = (const float*)d_in[5];
  const float* dt_bias  = (const float*)d_in[6];
  const float* A_log    = (const float*)d_in[7];
  const float* Dp       = (const float*)d_in[8];
  const float* norm_w   = (const float*)d_in[9];
  const float* W_out    = (const float*)d_in[10];
  const float* ln_w     = (const float*)d_in[11];
  const float* ln_b     = (const float*)d_in[12];
  const float* lo_w     = (const float*)d_in[13];
  const float* lo_b     = (const float*)d_in[14];

  char* ws = (char*)d_ws;
  size_t off = 0;
  auto alloc = [&](size_t bytes) {
    size_t o = off;
    off = (off + bytes + 255) & ~(size_t)255;
    return o;
  };
  float* h    = (float*)(ws + alloc((size_t)NTOK * D_MODEL * 4));
  float* zxb  = (float*)(ws + alloc((size_t)NTOK * D_PROJ  * 4));
  float* xBC  = (float*)(ws + alloc((size_t)NTOK * CONV_DIM* 4));
  float* dtv  = (float*)(ws + alloc((size_t)NTOK * NH      * 4));
  float* ybuf = (float*)(ws + alloc((size_t)NTOK * D_INNER * 4));
  float* mtmp = (float*)(ws + alloc((size_t)NTOK * D_MODEL * 4));
  (void)ws_size; (void)in_sizes; (void)n_in; (void)out_size;

  // linear_in: h = x @ lin_in_w + b    [32768 x 64] x [64 x 256]
  k_gemm<<<dim3(D_MODEL / 64, NTOK / 64), 128, 0, stream>>>(
      x, lin_in_w, h, lin_in_b, NTOK, D_MODEL, N_CH);

  for (int i = 0; i < NL; ++i) {
    // zxbcdt = h @ W_in[i]    [32768 x 256] x [256 x 1160]
    k_gemm<<<dim3((D_PROJ + 63) / 64, NTOK / 64), 128, 0, stream>>>(
        h, W_in + (size_t)i * D_MODEL * D_PROJ, zxb, nullptr, NTOK, D_PROJ, D_MODEL);
    // depthwise causal conv + SiLU
    k_conv_silu<<<(NTOK * CONV_DIM + 255) / 256, 256, 0, stream>>>(
        zxb, conv_w + (size_t)i * CONV_DIM * D_CONV, conv_b + (size_t)i * CONV_DIM, xBC);
    // dt = softplus(dt + dt_bias)
    k_dt<<<(NTOK * NH + 255) / 256, 256, 0, stream>>>(
        zxb, dt_bias + (size_t)i * NH, dtv);
    // chunked SSD scan -> ybuf (with D skip)
    k_scan<<<BATCH * NH, 128, 0, stream>>>(
        xBC, dtv, A_log + (size_t)i * NH, Dp + (size_t)i * NH, ybuf);
    // gated RMSNorm (in place on ybuf)
    k_gate_rms<<<NTOK, 128, 0, stream>>>(zxb, norm_w + (size_t)i * D_INNER, ybuf);
    // out proj: mtmp = ybuf @ W_out[i]   [32768 x 512] x [512 x 256]
    k_gemm<<<dim3(D_MODEL / 64, NTOK / 64), 128, 0, stream>>>(
        ybuf, W_out + (size_t)i * D_INNER * D_MODEL, mtmp, nullptr, NTOK, D_MODEL, D_INNER);
    // h = LN(mtmp + h)
    k_res_ln<<<NTOK, 128, 0, stream>>>(
        mtmp, ln_w + (size_t)i * D_MODEL, ln_b + (size_t)i * D_MODEL, h);
  }

  // linear_out
  k_linear_out<<<(NTOK * N_CLS + 255) / 256, 256, 0, stream>>>(h, lo_w, lo_b, (float*)d_out);
}
